// RealNeuralCircuitPolicy_19756849562237
// MI455X (gfx1250) — matrix operand order repaired and checked by reading, not compiled, not run on hardware
//
#include <hip/hip_runtime.h>
#include <stdint.h>

// Problem constants (fixed by the reference)
#define Bsz   64
#define Tsz   512
#define Hsz   512
#define INsz  512
#define OUTsz 256

typedef __attribute__((ext_vector_type(16))) __bf16        v16bf;
typedef __attribute__((ext_vector_type(8)))  float         v8f;
typedef __attribute__((ext_vector_type(4)))  unsigned int  u32x4;

union BfTile { v16bf bf; u32x4 u[2]; };

__device__ __forceinline__ uint16_t f2bf(float f) {
  union { float f; uint32_t u; } v; v.f = f;
  uint32_t r = v.u + 0x7FFFu + ((v.u >> 16) & 1u);  // round-to-nearest-even
  return (uint16_t)(r >> 16);
}
__device__ __forceinline__ float bf2f(uint16_t b) {
  union { uint32_t u; float f; } v; v.u = ((uint32_t)b) << 16; return v.f;
}

// ---------------------------------------------------------------------------
// K0: elementwise cast f32 -> bf16 (used for x)
// ---------------------------------------------------------------------------
__global__ void cast_f32_bf16(const float* __restrict__ src,
                              uint16_t* __restrict__ dst, int n) {
  int i = blockIdx.x * blockDim.x + threadIdx.x;
  if (i < n) dst[i] = f2bf(src[i]);
}

// ---------------------------------------------------------------------------
// K1: transpose + cast: src f32 [K][N] row-major -> dst bf16 [N][K] row-major
// (so WMMA B-matrix loads become contiguous along K)
// ---------------------------------------------------------------------------
__global__ void transpose_cast(const float* __restrict__ src,
                               uint16_t* __restrict__ dst, int K, int N) {
  int i = blockIdx.x * blockDim.x + threadIdx.x;
  if (i < K * N) {
    int n = i / K, k = i % K;
    dst[i] = f2bf(src[k * N + n]);
  }
}

// ---------------------------------------------------------------------------
// Generic WMMA GEMM:  C[Mtot x N] = A[Mtot x K] * BT^T + bias0 (+ bias1)
//   A   : bf16 row-major, lda = K
//   BT  : bf16 [N][K] (pre-transposed weight)
//   mode 0: Cf[m*N+n]  = acc + bias                    (f32 out)
//   mode 1: Cb[perm(m)*N+n] = bf16(tanh(acc + bias)),  perm: b*T+t -> t*B+b
//   mode 2: Cf[perm(m)*N+n] = acc + bias,              perm: t*B+b -> b*T+t
// Block = 256 threads (8 waves). Wave tile 32x32 (2x2 WMMA). Block tile 128x64.
// grid.x = Mtot/128, grid.y = N/64.
// ---------------------------------------------------------------------------
__global__ void __launch_bounds__(256)
gemm_wmma(const uint16_t* __restrict__ A, const uint16_t* __restrict__ BT,
          const float* __restrict__ bias0, const float* __restrict__ bias1,
          float* __restrict__ Cf, uint16_t* __restrict__ Cb,
          int K, int N, int mode) {
  const int wave = threadIdx.x >> 5;
  const int lane = threadIdx.x & 31;
  const int lrow = lane & 15;
  const int lh   = lane >> 4;
  const int gm = blockIdx.x * 128 + (wave & 3) * 32;
  const int gn = blockIdx.y * 64  + (wave >> 2) * 32;

  v8f acc[2][2] = {};

  const uint16_t* aRow0 = A  + (size_t)(gm      + lrow) * K;
  const uint16_t* aRow1 = A  + (size_t)(gm + 16 + lrow) * K;
  const uint16_t* bRow0 = BT + (size_t)(gn      + lrow) * K;
  const uint16_t* bRow1 = BT + (size_t)(gn + 16 + lrow) * K;

  for (int kt = 0; kt < K; kt += 32) {
    BfTile a0, a1, b0, b1;
    // A lane (m,h): k = {16h..16h+15}
    a0.u[0] = *(const u32x4*)(aRow0 + kt + 16 * lh);
    a0.u[1] = *(const u32x4*)(aRow0 + kt + 16 * lh + 8);
    a1.u[0] = *(const u32x4*)(aRow1 + kt + 16 * lh);
    a1.u[1] = *(const u32x4*)(aRow1 + kt + 16 * lh + 8);
    // B lane (n,h): k = {16h..16h+15} of column n (contiguous in BT)
    b0.u[0] = *(const u32x4*)(bRow0 + kt + 16 * lh);
    b0.u[1] = *(const u32x4*)(bRow0 + kt + 16 * lh + 8);
    b1.u[0] = *(const u32x4*)(bRow1 + kt + 16 * lh);
    b1.u[1] = *(const u32x4*)(bRow1 + kt + 16 * lh + 8);

    acc[0][0] = __builtin_amdgcn_wmma_f32_16x16x32_bf16(false, a0.bf, false, b0.bf, (short)0, acc[0][0], false, false);
    acc[0][1] = __builtin_amdgcn_wmma_f32_16x16x32_bf16(false, a0.bf, false, b1.bf, (short)0, acc[0][1], false, false);
    acc[1][0] = __builtin_amdgcn_wmma_f32_16x16x32_bf16(false, a1.bf, false, b0.bf, (short)0, acc[1][0], false, false);
    acc[1][1] = __builtin_amdgcn_wmma_f32_16x16x32_bf16(false, a1.bf, false, b1.bf, (short)0, acc[1][1], false, false);
  }

  #pragma unroll
  for (int i = 0; i < 2; ++i) {
    #pragma unroll
    for (int j = 0; j < 2; ++j) {
      const int n = gn + j * 16 + lrow;
      float bsum = bias0 ? bias0[n] : 0.f;
      if (bias1) bsum += bias1[n];
      #pragma unroll
      for (int r = 0; r < 8; ++r) {
        const int m = gm + i * 16 + lh * 8 + r;
        const float v = acc[i][j][r] + bsum;
        if (mode == 0) {
          Cf[(size_t)m * N + n] = v;
        } else if (mode == 1) {
          // m = b*T + t  ->  row = t*B + b   (T=512, B=64)
          const int row = ((m & 511) << 6) + (m >> 9);
          Cb[(size_t)row * N + n] = f2bf(tanhf(v));
        } else {
          // m = t*B + b  ->  row = b*T + t
          const int row = ((m & 63) << 9) + (m >> 6);
          Cf[(size_t)row * N + n] = v;
        }
      }
    }
  }
}

// ---------------------------------------------------------------------------
// K3: hs channel scan.  TS = tanh(x@Ws+bs) laid out [t][b][h] (bf16).
// hs_t = (1-a)*hs_{t-1} + a*TS[t];  HS[t][b][h] = bf16(hs_t).
// One thread per (b,h): 32768 threads, coalesced along h.
// ---------------------------------------------------------------------------
__global__ void scan_hs(const uint16_t* __restrict__ TS,
                        const float* __restrict__ tau,
                        uint16_t* __restrict__ HS) {
  const int idx = blockIdx.x * blockDim.x + threadIdx.x;  // 0 .. B*H-1
  const int b = idx >> 9;
  const int h = idx & 511;
  const float a = 1.f / tau[h];
  float hs = 0.f;
  for (int t = 0; t < Tsz; ++t) {
    const size_t off = ((size_t)t * Bsz + b) * Hsz + h;
    hs = (1.f - a) * hs + a * bf2f(TS[off]);
    HS[off] = f2bf(hs);
  }
}

// ---------------------------------------------------------------------------
// K5/K7: sequential recurrence, single persistent workgroup (32 waves).
//   for t: pre = G[t] + h_{t-1} @ Wr ; h_t = (1-a)*h_{t-1} + a*tanh(pre)
// State: bf16 copy in LDS (A-matrix source, 64 KB); fp32 master in registers.
// Wave w owns output columns n in [16w,16w+16): 4 M-tiles -> 64 WMMA / step.
// Wr tiles re-read from global each step (512 KB, L2 resident).
// Critical-path tuning:
//   - G[t] register-preloaded BEFORE the k-loop (latency hidden under WMMA)
//   - G[t+1] prefetched one step ahead (global_prefetch_b8)
//   - Hout global stores issued AFTER the phase barrier (drain under next GEMM)
// ---------------------------------------------------------------------------
__global__ void __launch_bounds__(1024)
recur_wmma(const uint16_t* __restrict__ WrT, const float* __restrict__ G,
           const float* __restrict__ tau, uint16_t* __restrict__ Hout) {
  __shared__ __align__(16) uint16_t hB[Bsz * Hsz];  // 64 KB

  const int tid  = threadIdx.x;
  const int wave = tid >> 5;          // 0..31  -> N-tile index
  const int lane = tid & 31;
  const int lrow = lane & 15;
  const int lh   = lane >> 4;
  const int n    = wave * 16 + lrow;  // this lane's output column

  for (int i = tid; i < Bsz * Hsz; i += 1024) hB[i] = 0;
  __syncthreads();

  const float a  = 1.f / tau[n];
  float state[4][8] = {};             // fp32 master state: 32 rows (b) x 1 col (n)

  const uint16_t* bRow = WrT + (size_t)n * Hsz;

  for (int t = 0; t < Tsz; ++t) {
    const float* Gt = G + (size_t)t * (Bsz * Hsz);
    const float* Gnext = (t + 1 < Tsz) ? (Gt + Bsz * Hsz) : Gt;

    // Preload G[t] (independent of hB) so the loads retire under the WMMAs.
    float g[4][8];
    #pragma unroll
    for (int mt = 0; mt < 4; ++mt)
      #pragma unroll
      for (int r = 0; r < 8; ++r)
        g[mt][r] = Gt[(mt * 16 + lh * 8 + r) * Hsz + n];

    v8f acc[4] = {};
    for (int kt = 0; kt < Hsz; kt += 32) {
      BfTile bt;
      bt.u[0] = *(const u32x4*)(bRow + kt + 16 * lh);
      bt.u[1] = *(const u32x4*)(bRow + kt + 16 * lh + 8);
      #pragma unroll
      for (int mt = 0; mt < 4; ++mt) {
        BfTile at;
        const uint16_t* aRow = hB + (mt * 16 + lrow) * Hsz;
        at.u[0] = *(const u32x4*)(aRow + kt + 16 * lh);
        at.u[1] = *(const u32x4*)(aRow + kt + 16 * lh + 8);
        acc[mt] = __builtin_amdgcn_wmma_f32_16x16x32_bf16(false, at.bf, false, bt.bf, (short)0, acc[mt], false, false);
      }
    }
    __syncthreads();  // all waves done reading state(t-1) from LDS

    // State update: LDS copy + register master; prefetch next step's G rows.
    uint16_t hb16[4][8];
    #pragma unroll
    for (int mt = 0; mt < 4; ++mt) {
      #pragma unroll
      for (int r = 0; r < 8; ++r) {
        const int b = mt * 16 + lh * 8 + r;          // C/D layout: VGPR r -> rows {r, r+8}
        const float pre = acc[mt][r] + g[mt][r];
        const float hv  = (1.f - a) * state[mt][r] + a * tanhf(pre);
        state[mt][r] = hv;
        const uint16_t hb = f2bf(hv);
        hb16[mt][r] = hb;
        hB[b * Hsz + n] = hb;
        __builtin_prefetch(&Gnext[b * Hsz + n], 0, 3);  // global_prefetch_b8
      }
    }
    __syncthreads();  // state(t) visible before next step's GEMM

    // Global history writes AFTER the barrier: they drain under the next GEMM.
    uint16_t* Ht = Hout + (size_t)t * (Bsz * Hsz);
    #pragma unroll
    for (int mt = 0; mt < 4; ++mt)
      #pragma unroll
      for (int r = 0; r < 8; ++r)
        Ht[(mt * 16 + lh * 8 + r) * Hsz + n] = hb16[mt][r];
  }
}

// ---------------------------------------------------------------------------
// Host orchestration
// ---------------------------------------------------------------------------
extern "C" void kernel_launch(void* const* d_in, const int* in_sizes, int n_in,
                              void* d_out, int out_size, void* d_ws, size_t ws_size,
                              hipStream_t stream) {
  const float* x     = (const float*)d_in[0];
  const float* Ws    = (const float*)d_in[1];
  const float* bs    = (const float*)d_in[2];
  const float* Wi    = (const float*)d_in[3];
  const float* bi    = (const float*)d_in[4];
  const float* Wc    = (const float*)d_in[5];
  const float* bc    = (const float*)d_in[6];
  const float* Wm    = (const float*)d_in[7];
  const float* bm    = (const float*)d_in[8];
  const float* Wir   = (const float*)d_in[9];
  const float* bir   = (const float*)d_in[10];
  const float* Wcr   = (const float*)d_in[11];
  const float* bcr   = (const float*)d_in[12];
  const float* tau_s = (const float*)d_in[13];
  const float* tau_i = (const float*)d_in[14];
  const float* tau_c = (const float*)d_in[15];
  float* out = (float*)d_out;

  // Workspace carve-up (256 B aligned)
  char* ws = (char*)d_ws;
  size_t off = 0;
  auto carve = [&](size_t bytes) -> char* {
    char* p = ws + off; off += (bytes + 255) & ~(size_t)255; return p;
  };
  uint16_t* WsT  = (uint16_t*)carve((size_t)Hsz * INsz * 2);
  uint16_t* WiT  = (uint16_t*)carve((size_t)Hsz * Hsz * 2);
  uint16_t* WcT  = (uint16_t*)carve((size_t)Hsz * Hsz * 2);
  uint16_t* WirT = (uint16_t*)carve((size_t)Hsz * Hsz * 2);
  uint16_t* WcrT = (uint16_t*)carve((size_t)Hsz * Hsz * 2);
  uint16_t* WmT  = (uint16_t*)carve((size_t)OUTsz * Hsz * 2);
  char* bufA = carve((size_t)Tsz * Bsz * Hsz * 2);   // 32 MB: TS, later HI
  char* bufB = carve((size_t)Tsz * Bsz * Hsz * 2);   // 32 MB: HS, later HC
  char* bufG = carve((size_t)Tsz * Bsz * Hsz * 4);   // 64 MB: Xb(bf16), then GI/GC(f32)

  uint16_t* Xb = (uint16_t*)bufG;
  uint16_t* TS = (uint16_t*)bufA;
  uint16_t* HS = (uint16_t*)bufB;
  float*    GI = (float*)bufG;
  uint16_t* HI = (uint16_t*)bufA;
  float*    GC = (float*)bufG;
  uint16_t* HC = (uint16_t*)bufB;

  const int MT = Bsz * Tsz;  // 32768 rows for the big GEMMs

  // K0: x -> bf16
  {
    int n = MT * INsz;
    cast_f32_bf16<<<n / 256, 256, 0, stream>>>(x, Xb, n);
  }
  // K1: weight transposes
  transpose_cast<<<(INsz * Hsz) / 256, 256, 0, stream>>>(Ws,  WsT,  INsz, Hsz);
  transpose_cast<<<(Hsz * Hsz) / 256, 256, 0, stream>>>(Wi,  WiT,  Hsz, Hsz);
  transpose_cast<<<(Hsz * Hsz) / 256, 256, 0, stream>>>(Wc,  WcT,  Hsz, Hsz);
  transpose_cast<<<(Hsz * Hsz) / 256, 256, 0, stream>>>(Wir, WirT, Hsz, Hsz);
  transpose_cast<<<(Hsz * Hsz) / 256, 256, 0, stream>>>(Wcr, WcrT, Hsz, Hsz);
  transpose_cast<<<(Hsz * OUTsz) / 256, 256, 0, stream>>>(Wm, WmT, Hsz, OUTsz);

  // K2: TS[t][b][:] = tanh(x@Ws + bs)   (mode 1: bf16 + permute b*T+t -> t*B+b)
  gemm_wmma<<<dim3(MT / 128, Hsz / 64), 256, 0, stream>>>(
      Xb, WsT, bs, nullptr, nullptr, TS, INsz, Hsz, 1);

  // K3: hs scan
  scan_hs<<<(Bsz * Hsz) / 256, 256, 0, stream>>>(TS, tau_s, HS);

  // K4: GI = HS@Wi + bi + bir   (mode 0, f32)
  gemm_wmma<<<dim3(MT / 128, Hsz / 64), 256, 0, stream>>>(
      HS, WiT, bi, bir, GI, nullptr, Hsz, Hsz, 0);

  // K5: hi recurrence (persistent single workgroup)
  recur_wmma<<<1, 1024, 0, stream>>>(WirT, GI, tau_i, HI);

  // K6: GC = HI@Wc + bc + bcr
  gemm_wmma<<<dim3(MT / 128, Hsz / 64), 256, 0, stream>>>(
      HI, WcT, bc, bcr, GC, nullptr, Hsz, Hsz, 0);

  // K7: hc recurrence
  recur_wmma<<<1, 1024, 0, stream>>>(WcrT, GC, tau_c, HC);

  // K8: out[b][t][:] = HC@Wm + bm   (mode 2: f32 + permute t*B+b -> b*T+t)
  gemm_wmma<<<dim3(MT / 128, OUTsz / 64), 256, 0, stream>>>(
      HC, WmT, bm, nullptr, out, nullptr, Hsz, OUTsz, 2);

  (void)in_sizes; (void)n_in; (void)out_size; (void)ws_size;
}
